// GAT2Layer_5643587027338
// MI455X (gfx1250) — matrix operand order repaired and checked by reading, not compiled, not run on hardware
//
#include <hip/hip_runtime.h>

// ---------------------------------------------------------------------------
// 3-layer GAT on gfx1250 (MI455X).
//  * GEMMs run on the WMMA pipe: v_wmma_f32_16x16x32_bf16 (fp32 accum).
//  * Segment softmax: orderable-uint atomicMax + atomicAdd(f32) in L2.
//  * Whole working set (~70 MB) fits in the 192 MB L2.
// ---------------------------------------------------------------------------

typedef __bf16 v16bf __attribute__((ext_vector_type(16)));
typedef float  v8f   __attribute__((ext_vector_type(8)));

#define N_NODES 50000
#define N_EDGES 800000
#define N_EP    (N_EDGES + N_NODES)
#define NEG_SLOPE 0.2f
#define EPS_SM 1e-16f
#define ORD_NEG_INF 0x007FFFFFu   // orderable encoding of -inf

__device__ __forceinline__ unsigned short bf16_rne(float f) {
  unsigned b = __float_as_uint(f);
  return (unsigned short)((b + 0x7FFFu + ((b >> 16) & 1u)) >> 16);
}
__device__ __forceinline__ unsigned ord_enc(float f) {
  unsigned b = __float_as_uint(f);
  return (b & 0x80000000u) ? ~b : (b | 0x80000000u);
}
__device__ __forceinline__ float ord_dec(unsigned o) {
  unsigned b = (o & 0x80000000u) ? (o & 0x7FFFFFFFu) : ~o;
  return __uint_as_float(b);
}

// --- edge list with self loops; also emits the int64 [2,E'] output ----------
__global__ void k_build_edges(const long long* __restrict__ ei,
                              int* __restrict__ src32, int* __restrict__ dst32,
                              long long* __restrict__ out_edges) {
  int k = blockIdx.x * blockDim.x + threadIdx.x;
  if (k >= N_EP) return;
  int s, d;
  if (k < N_EDGES) { s = (int)ei[k]; d = (int)ei[N_EDGES + k]; }
  else             { s = k - N_EDGES; d = s; }
  src32[k] = s; dst32[k] = d;
  out_edges[k] = (long long)s;
  out_edges[(size_t)N_EP + k] = (long long)d;
}

__global__ void k_f32_to_bf16(const float* __restrict__ in,
                              unsigned short* __restrict__ out, int n) {
  int i = blockIdx.x * blockDim.x + threadIdx.x;
  if (i < n) out[i] = bf16_rne(in[i]);
}

__global__ void k_zero_f32(float* __restrict__ p, int n) {
  int i = blockIdx.x * blockDim.x + threadIdx.x;
  if (i < n) p[i] = 0.f;
}

__global__ void k_init_nodes(unsigned* __restrict__ mu, float* __restrict__ s, int n) {
  int i = blockIdx.x * blockDim.x + threadIdx.x;
  if (i < n) { mu[i] = ORD_NEG_INF; s[i] = 0.f; }
}

// --- WMMA GEMM: h[N,FOUT] = xb[N,FIN] @ wb[FOUT,FIN]^T  (bf16 in, f32 out) --
// One wave computes one 16x16 output tile. Per-lane fragment packing follows
// the CDNA5 ISA 16-bit A(16x32)/B(32x16)/C(16x16) VGPR layouts; packed bf16
// pairs are K-contiguous in memory so each is a single dword load.
template <int FIN, int FOUT>
__global__ __launch_bounds__(256)
void k_gemm_wmma(const unsigned short* __restrict__ xb,
                 const unsigned short* __restrict__ wb,
                 float* __restrict__ h, int n_tiles) {
  const int lane = threadIdx.x & 31;
  const int wave = threadIdx.x >> 5;
  const int tile = blockIdx.x * (blockDim.x >> 5) + wave;
  if (tile >= n_tiles) return;                   // wave-uniform: EXEC stays all-1s
  constexpr int CT = FOUT / 16;
  const int rowTile = tile / CT;
  const int colTile = tile % CT;
  const int r  = lane & 15;
  const int hi = lane >> 4;

  const unsigned short* xrow = xb + (size_t)(rowTile * 16 + r) * FIN;
  const unsigned short* wrow = wb + (size_t)(colTile * 16 + r) * FIN;

  v8f acc = {};
  for (int k0 = 0; k0 < FIN; k0 += 32) {
    union Frag { unsigned u[8]; v16bf v; } A, B;
#pragma unroll
    for (int p = 0; p < 8; ++p) {
      // A 16x32: lane r row M=r; VGPR p holds K pair {(p<4?0:16)+hi*8+(p&3)*2, +1}
      const int ka = k0 + ((p < 4) ? 0 : 16) + hi * 8 + (p & 3) * 2;
      A.u[p] = *(const unsigned*)(xrow + ka);
      // B 32x16: lane r col N=r; VGPR p holds K pair {(p<4?0:16)+hi*8+(p&3)*2, +1}
      const int kb = k0 + ((p < 4) ? 0 : 16) + hi * 8 + (p & 3) * 2;
      B.u[p] = *(const unsigned*)(wrow + kb);
    }
    acc = __builtin_amdgcn_wmma_f32_16x16x32_bf16(
        /*neg_a=*/false, A.v, /*neg_b=*/false, B.v,
        /*c_mod=*/(short)0, acc, /*reuse_a=*/false, /*reuse_b=*/false);
  }
  // D 16x16 f32: VGPR i -> row M = hi*8+i, col N = r
  float* hout = h + (size_t)rowTile * 16 * FOUT + colTile * 16 + r;
#pragma unroll
  for (int i = 0; i < 8; ++i)
    hout[(size_t)(hi * 8 + i) * FOUT] = acc[i];
}

// --- per-node attention scores: as = h.a_src, ad = h.a_dst ------------------
template <int F>
__global__ void k_scores(const float* __restrict__ h,
                         const float* __restrict__ a_s, const float* __restrict__ a_d,
                         float* __restrict__ asn, float* __restrict__ adn, int n) {
  int i = blockIdx.x * blockDim.x + threadIdx.x;
  if (i >= n) return;
  const float* row = h + (size_t)i * F;
  float s = 0.f, d = 0.f;
#pragma unroll 8
  for (int f = 0; f < F; ++f) { float v = row[f]; s += v * a_s[f]; d += v * a_d[f]; }
  asn[i] = s; adn[i] = d;
}

// --- e = leaky_relu(as[src]+ad[dst]); segment max via atomicMax(u32) --------
__global__ void k_edge_logits(const int* __restrict__ src, const int* __restrict__ dst,
                              const float* __restrict__ asn, const float* __restrict__ adn,
                              float* __restrict__ e, unsigned* __restrict__ mu, int ne) {
  int k = blockIdx.x * blockDim.x + threadIdx.x;
  if (k >= ne) return;
  float v = asn[src[k]] + adn[dst[k]];
  v = v > 0.f ? v : NEG_SLOPE * v;
  e[k] = v;
  atomicMax(mu + dst[k], ord_enc(v));
}

// --- ex = exp(e - m[dst]); segment sum --------------------------------------
__global__ void k_edge_exp(const int* __restrict__ dst, float* __restrict__ e,
                           const unsigned* __restrict__ mu, float* __restrict__ s, int ne) {
  int k = blockIdx.x * blockDim.x + threadIdx.x;
  if (k >= ne) return;
  float v = __expf(e[k] - ord_dec(mu[dst[k]]));
  e[k] = v;
  atomicAdd(s + dst[k], v);
}

// --- alpha = ex / (s[dst]+eps); optionally export (layer-1 output) ----------
__global__ void k_edge_alpha(const int* __restrict__ dst, float* __restrict__ e,
                             const float* __restrict__ s, float* __restrict__ alpha_out, int ne) {
  int k = blockIdx.x * blockDim.x + threadIdx.x;
  if (k >= ne) return;
  float a = e[k] / (s[dst[k]] + EPS_SM);
  e[k] = a;
  if (alpha_out) alpha_out[k] = a;
}

// --- acc[dst] += alpha * h[src]; 16 threads/edge for F=64 (coalesced feats) -
template <int F>
__global__ void k_aggregate(const int* __restrict__ src, const int* __restrict__ dst,
                            const float* __restrict__ alpha, const float* __restrict__ h,
                            float* __restrict__ acc, int ne) {
  constexpr int FC = F / 4;
  int t = blockIdx.x * blockDim.x + threadIdx.x;
  int k = t / FC;
  if (k >= ne) return;
  int c = (t % FC) * 4;
  float a = alpha[k];
  const float4 hv = *(const float4*)(h + (size_t)src[k] * F + c);
  float* o = acc + (size_t)dst[k] * F + c;
  atomicAdd(o + 0, a * hv.x);
  atomicAdd(o + 1, a * hv.y);
  atomicAdd(o + 2, a * hv.z);
  atomicAdd(o + 3, a * hv.w);
}

// --- out = acc + b [, relu]; emit f32 (final) and/or bf16 (next-layer GEMM) -
template <int F, bool RELU>
__global__ void k_finalize(const float* __restrict__ acc, const float* __restrict__ bias,
                           float* __restrict__ outf, unsigned short* __restrict__ outb, int n) {
  int t = blockIdx.x * blockDim.x + threadIdx.x;
  if (t >= n * F) return;
  float v = acc[t] + bias[t % F];
  if (RELU) v = v > 0.f ? v : 0.f;
  if (outf) outf[t] = v;
  if (outb) outb[t] = bf16_rne(v);
}

// ---------------------------------------------------------------------------
template <int FIN, int FOUT, bool RELU>
static void run_layer(const unsigned short* xin_b, const float* W,
                      const float* a_s, const float* a_d, const float* bias,
                      float* alpha_out, float* feat_out, unsigned short* next_b,
                      const int* src32, const int* dst32, unsigned short* wb,
                      float* h, float* acc, float* asn, float* adn,
                      unsigned* mu, float* ssum, float* ebuf, hipStream_t st) {
  constexpr int TPB = 256;
  k_f32_to_bf16<<<(FOUT * FIN + TPB - 1) / TPB, TPB, 0, st>>>(W, wb, FOUT * FIN);
  const int tiles = (N_NODES / 16) * (FOUT / 16);
  k_gemm_wmma<FIN, FOUT><<<(tiles + 7) / 8, TPB, 0, st>>>(xin_b, wb, h, tiles);
  k_scores<FOUT><<<(N_NODES + TPB - 1) / TPB, TPB, 0, st>>>(h, a_s, a_d, asn, adn, N_NODES);
  k_init_nodes<<<(N_NODES + TPB - 1) / TPB, TPB, 0, st>>>(mu, ssum, N_NODES);
  k_zero_f32<<<(N_NODES * FOUT + TPB - 1) / TPB, TPB, 0, st>>>(acc, N_NODES * FOUT);
  k_edge_logits<<<(N_EP + TPB - 1) / TPB, TPB, 0, st>>>(src32, dst32, asn, adn, ebuf, mu, N_EP);
  k_edge_exp<<<(N_EP + TPB - 1) / TPB, TPB, 0, st>>>(dst32, ebuf, mu, ssum, N_EP);
  k_edge_alpha<<<(N_EP + TPB - 1) / TPB, TPB, 0, st>>>(dst32, ebuf, ssum, alpha_out, N_EP);
  const int aggT = N_EP * (FOUT / 4);
  k_aggregate<FOUT><<<(aggT + TPB - 1) / TPB, TPB, 0, st>>>(src32, dst32, ebuf, h, acc, N_EP);
  k_finalize<FOUT, RELU><<<(N_NODES * FOUT + TPB - 1) / TPB, TPB, 0, st>>>(acc, bias, feat_out, next_b, N_NODES);
}

extern "C" void kernel_launch(void* const* d_in, const int* in_sizes, int n_in,
                              void* d_out, int out_size, void* d_ws, size_t ws_size,
                              hipStream_t stream) {
  const float*     x   = (const float*)d_in[0];
  const long long* ei  = (const long long*)d_in[1];   // int64 edge_index [2,E]
  const float* W1  = (const float*)d_in[2];
  const float* as1 = (const float*)d_in[3];
  const float* ad1 = (const float*)d_in[4];
  const float* b1  = (const float*)d_in[5];
  const float* W2  = (const float*)d_in[6];
  const float* as2 = (const float*)d_in[7];
  const float* ad2 = (const float*)d_in[8];
  const float* b2  = (const float*)d_in[9];
  const float* W3  = (const float*)d_in[10];
  const float* as3 = (const float*)d_in[11];
  const float* ad3 = (const float*)d_in[12];
  const float* b3  = (const float*)d_in[13];

  // workspace carve-out (~56 MB total)
  char* ws = (char*)d_ws;
  size_t cur = 0;
  auto take = [&](size_t bytes) -> void* {
    void* p = ws + cur;
    cur = (cur + bytes + 255) & ~(size_t)255;
    return p;
  };
  int*            src32 = (int*)take((size_t)N_EP * 4);
  int*            dst32 = (int*)take((size_t)N_EP * 4);
  unsigned short* xb    = (unsigned short*)take((size_t)N_NODES * 128 * 2);
  unsigned short* xb2   = (unsigned short*)take((size_t)N_NODES * 64 * 2);
  unsigned short* wb    = (unsigned short*)take((size_t)64 * 128 * 2);
  float*          h     = (float*)take((size_t)N_NODES * 64 * 4);
  float*          acc   = (float*)take((size_t)N_NODES * 64 * 4);
  float*          asn   = (float*)take((size_t)N_NODES * 4);
  float*          adn   = (float*)take((size_t)N_NODES * 4);
  unsigned*       mu    = (unsigned*)take((size_t)N_NODES * 4);
  float*          ssum  = (float*)take((size_t)N_NODES * 4);
  float*          ebuf  = (float*)take((size_t)N_EP * 4);

  // output layout: [2*E' int64][E' f32 alpha1][N*32 f32 out], byte-contiguous
  long long* out_edges = (long long*)d_out;
  float*     out_alpha = (float*)((char*)d_out + (size_t)2 * N_EP * 8);
  float*     out_feat  = out_alpha + N_EP;

  constexpr int TPB = 256;
  k_build_edges<<<(N_EP + TPB - 1) / TPB, TPB, 0, stream>>>(ei, src32, dst32, out_edges);
  k_f32_to_bf16<<<(N_NODES * 128 + TPB - 1) / TPB, TPB, 0, stream>>>(x, xb, N_NODES * 128);

  run_layer<128, 64, true >(xb,  W1, as1, ad1, b1, out_alpha, nullptr,  xb2,
                            src32, dst32, wb, h, acc, asn, adn, mu, ssum, ebuf, stream);
  run_layer<64,  64, true >(xb2, W2, as2, ad2, b2, nullptr,   nullptr,  xb2,
                            src32, dst32, wb, h, acc, asn, adn, mu, ssum, ebuf, stream);
  run_layer<64,  32, false>(xb2, W3, as3, ad3, b3, nullptr,   out_feat, nullptr,
                            src32, dst32, wb, h, acc, asn, adn, mu, ssum, ebuf, stream);
}
